// MemoryModule3D_25701084299676
// MI455X (gfx1250) — matrix operand order repaired and checked by reading, not compiled, not run on hardware
//
#include <hip/hip_runtime.h>
#include <hip/hip_fp16.h>

typedef _Float16 v16h __attribute__((ext_vector_type(16)));
typedef float    v8f  __attribute__((ext_vector_type(8)));
typedef float    f4   __attribute__((ext_vector_type(4)));

#define T_TOK   32768
#define C_DIM   256
#define M_DIM   2000
#define NT1     125       // 2000/16 N-tiles in GEMM1
#define KT1     8         // 256/32  K-tiles in GEMM1
#define MT      2048      // 32768/16 token tiles
#define KT2     64        // 2048/32 K-tiles in GEMM2 (K padded 2000->2048)
#define LSTRIDE 2052      // f32 row stride for logit panel (stride%8==4 -> conflict-free col reads)
#define ZSTRIDE 257       // odd stride -> conflict-free transpose reads

__device__ __forceinline__ float wred_max(float v){
#pragma unroll
  for (int o = 16; o > 0; o >>= 1) v = fmaxf(v, __shfl_xor(v, o, 32));
  return v;
}
__device__ __forceinline__ float wred_sum(float v){
#pragma unroll
  for (int o = 16; o > 0; o >>= 1) v += __shfl_xor(v, o, 32);
  return v;
}
__device__ __forceinline__ v16h ldfrag(const _Float16* p){
  union { f4 f[2]; v16h h; } u;
  u.f[0] = *(const f4*)p;
  u.f[1] = *(const f4*)(p + 8);
  return u.h;
}

// ---- Prep: per-token inverse L2 norm of z (lanes ride token dim -> coalesced) ----
__global__ void k_znorm(const float* __restrict__ z, float* __restrict__ invZ){
  int tok = blockIdx.x * blockDim.x + threadIdx.x;
  if (tok >= T_TOK) return;
  int nb = tok >> 12, rem = tok & 4095;
  const float* base = z + (size_t)nb * C_DIM * 4096 + rem;
  float s = 0.f;
  for (int c = 0; c < C_DIM; ++c) { float v = base[(size_t)c * 4096]; s += v * v; }
  invZ[tok] = 1.f / fmaxf(sqrtf(s), 1e-8f);
}

// ---- Prep: per-memory-row inverse L2 norm (one wave per row) ----
__global__ void k_mnorm(const float* __restrict__ mem, float* __restrict__ invM){
  int row = blockIdx.x * 8 + (threadIdx.x >> 5);
  int l = threadIdx.x & 31;
  if (row >= M_DIM) return;
  const float* p = mem + (size_t)row * C_DIM;
  float s = 0.f;
#pragma unroll
  for (int i = 0; i < 8; ++i) { float v = p[l + i * 32]; s += v * v; }
  s = wred_sum(s);
  if (l == 0) invM[row] = 1.f / fmaxf(sqrtf(s), 1e-8f);
}

// ---- Prep: pack B1 = normalized-memory^T (K=256 x N=2000) into WMMA-fragment order ----
// B 32x16 layout: lane half h holds K=kt*32+h*8+0..7 and kt*32+16+h*8+0..7 of column (l&15).
__global__ void k_packB1(const float* __restrict__ mem, const float* __restrict__ invM,
                         _Float16* __restrict__ B1p){
  int frag = blockIdx.x * 8 + (threadIdx.x >> 5);   // 1000 frags = 125 nt * 8 kt
  int l = threadIdx.x & 31;
  int nt = frag / KT1, kt = frag % KT1;
  int nIdx = nt * 16 + (l & 15);
  int kb = kt * 32 + (l >> 4) * 8;
  float inv = invM[nIdx];
  const float* src = mem + (size_t)nIdx * C_DIM;
  _Float16* dst = B1p + (size_t)frag * 512 + l * 16;
#pragma unroll
  for (int i = 0; i < 16; ++i) dst[i] = (_Float16)(src[kb + i + ((i >> 3) << 3)] * inv);
}

// ---- Prep: pack B2 = memory (K=2000->2048 zero-padded, N=256) into fragment order ----
__global__ void k_packB2(const float* __restrict__ mem, _Float16* __restrict__ B2p){
  int frag = blockIdx.x * 8 + (threadIdx.x >> 5);   // 1024 frags = 16 nt * 64 kt
  int l = threadIdx.x & 31;
  int nt = frag >> 6, kt = frag & 63;
  int nIdx = nt * 16 + (l & 15);
  int kb = kt * 32 + (l >> 4) * 8;
  _Float16* dst = B2p + (size_t)frag * 512 + l * 16;
#pragma unroll
  for (int i = 0; i < 16; ++i) {
    int k = kb + i + ((i >> 3) << 3);
    dst[i] = (k < M_DIM) ? (_Float16)mem[(size_t)k * C_DIM + nIdx] : (_Float16)0.f;
  }
}

// ---- Prep: pack A = normalized z tokens into fragment order (A 16x32 interleaved K-halves) ----
// Per ISA: lane half h holds K = h*8+0..7 and 16+h*8+0..7 across 8 VGPR pairs.
__global__ void k_packA(const float* __restrict__ z, const float* __restrict__ invZ,
                        _Float16* __restrict__ Ap){
  int mTile = blockIdx.x;                 // 2048 blocks
  int kt = threadIdx.x >> 5;              // 8 waves = 8 K-tiles
  int l = threadIdx.x & 31;
  int tok = mTile * 16 + (l & 15);
  int nb = tok >> 12, rem = tok & 4095;
  float inv = invZ[tok];
  int kb = kt * 32 + (l >> 4) * 8;
  const float* base = z + (size_t)nb * C_DIM * 4096 + rem;
  _Float16* dst = Ap + ((size_t)mTile * KT1 + kt) * 512 + l * 16;
#pragma unroll
  for (int i = 0; i < 8; ++i) dst[i]     = (_Float16)(base[(size_t)(kb + i) * 4096] * inv);
#pragma unroll
  for (int i = 0; i < 8; ++i) dst[8 + i] = (_Float16)(base[(size_t)(kb + 16 + i) * 4096] * inv);
}

// ---- Fused: GEMM1 -> softmax/shrink/L1 (LDS-resident) -> GEMM2 -> transposed store ----
__global__ __launch_bounds__(256, 1)
void k_main(const _Float16* __restrict__ Ap, const _Float16* __restrict__ B1p,
            const _Float16* __restrict__ B2p,
            float* __restrict__ zout, float* __restrict__ what){
  __shared__ float sW[16][LSTRIDE];   // 16 tokens x 2048(+pad) logits/weights, ~128 KB
  __shared__ float sZ[16][ZSTRIDE];   // 16 tokens x 256 output staging

  const int mTile = blockIdx.x;
  const int wave = threadIdx.x >> 5;
  const int l = threadIdx.x & 31;
  const int m0 = (l >> 4) * 8;
  const int nlo = l & 15;

  // Preload this tile's A fragments once (reused across all 125 N-tiles).
  v16h a[KT1];
#pragma unroll
  for (int kt = 0; kt < KT1; ++kt)
    a[kt] = ldfrag(Ap + ((size_t)mTile * KT1 + kt) * 512 + l * 16);

  // GEMM1: logits = zn @ mn^T, accumulated in f32 fragments, spilled to LDS.
  for (int nt = wave; nt < NT1; nt += 8) {   // wave-uniform bounds: EXEC stays all-ones
    v8f acc = {0.f, 0.f, 0.f, 0.f, 0.f, 0.f, 0.f, 0.f};
    const _Float16* bp = B1p + (size_t)nt * KT1 * 512 + l * 16;
#pragma unroll
    for (int kt = 0; kt < KT1; ++kt) {
      v16h b = ldfrag(bp + kt * 512);
      acc = __builtin_amdgcn_wmma_f32_16x16x32_f16(false, a[kt], false, b,
                                                   (short)0, acc, false, false);
    }
#pragma unroll
    for (int r = 0; r < 8; ++r) sW[m0 + r][nt * 16 + nlo] = acc[r];
  }
  __syncthreads();

  // Softmax + hard_shrink_relu + L1 normalize: each wave owns 2 of the 16 rows.
#pragma unroll
  for (int rr = 0; rr < 2; ++rr) {
    int row = wave * 2 + rr;
    float mx = -3.4e38f;
    for (int j = l; j < M_DIM; j += 32) mx = fmaxf(mx, sW[row][j]);
    mx = wred_max(mx);
    float s = 0.f;
    for (int j = l; j < M_DIM; j += 32) {
      float e = __expf(sW[row][j] - mx); sW[row][j] = e; s += e;
    }
    s = wred_sum(s);
    float invs = 1.f / s;
    float l1 = 0.f;
    for (int j = l; j < M_DIM; j += 32) {
      float w = sW[row][j] * invs;
      float t = w - 5e-4f;                       // LAM/MEM_DIM
      float wh = (t > 0.f) ? (t * w / (t + 1e-15f)) : 0.f;
      sW[row][j] = wh; l1 += wh;
    }
    l1 = wred_sum(l1);
    float sc = 1.f / fmaxf(l1, 1e-12f);
    float* wout = what + (size_t)(mTile * 16 + row) * M_DIM;
    for (int j = l; j < M_DIM; j += 32) {
      float wv = sW[row][j] * sc; sW[row][j] = wv; wout[j] = wv;  // coalesced w_hat store
    }
    for (int j = M_DIM + l; j < 2048; j += 32) sW[row][j] = 0.f;  // zero K padding
  }
  __syncthreads();

  // GEMM2: z_hat = w_hat @ memory. A comes from LDS (f32 -> f16), B pre-packed.
#pragma unroll
  for (int p = 0; p < 2; ++p) {
    int nt2 = wave + p * 8;                      // 16 N-tiles over 8 waves
    v8f acc = {0.f, 0.f, 0.f, 0.f, 0.f, 0.f, 0.f, 0.f};
    const int kh = (l >> 4) * 8;
    for (int kt = 0; kt < KT2; ++kt) {
      const float* wr = &sW[nlo][kt * 32 + kh];  // stride%8==4 -> conflict-free b128 reads
      f4 w0 = *(const f4*)(wr);
      f4 w1 = *(const f4*)(wr + 4);
      f4 w2 = *(const f4*)(wr + 16);
      f4 w3 = *(const f4*)(wr + 20);
      v16h af;
#pragma unroll
      for (int i = 0; i < 4; ++i) {
        af[i]      = (_Float16)w0[i];
        af[4 + i]  = (_Float16)w1[i];
        af[8 + i]  = (_Float16)w2[i];
        af[12 + i] = (_Float16)w3[i];
      }
      v16h b = ldfrag(B2p + ((size_t)nt2 * KT2 + kt) * 512 + l * 16);
      acc = __builtin_amdgcn_wmma_f32_16x16x32_f16(false, af, false, b,
                                                   (short)0, acc, false, false);
    }
#pragma unroll
    for (int r = 0; r < 8; ++r) sZ[m0 + r][nt2 * 16 + nlo] = acc[r];
  }
  __syncthreads();

  // Transposed store to NCDHW: lanes ride the token dim -> contiguous 64B chunks.
  int tok0 = mTile * 16;
  int nb = tok0 >> 12, rem0 = tok0 & 4095;
  int m = threadIdx.x & 15;
  int cb = threadIdx.x >> 4;
  float* ob = zout + (size_t)nb * C_DIM * 4096 + rem0 + m;
  for (int c = cb; c < C_DIM; c += 16) ob[(size_t)c * 4096] = sZ[m][c];
}

extern "C" void kernel_launch(void* const* d_in, const int* in_sizes, int n_in,
                              void* d_out, int out_size, void* d_ws, size_t ws_size,
                              hipStream_t stream) {
  const float* z   = (const float*)d_in[0];   // (8,256,4,32,32) f32
  const float* mem = (const float*)d_in[1];   // (2000,256) f32
  float* zout = (float*)d_out;                                    // 8*256*4*32*32
  float* what = zout + (size_t)8 * 256 * 4 * 32 * 32;             // 32768*2000

  uint8_t* ws = (uint8_t*)d_ws;
  float*    invZ = (float*)ws;                                     // 131072 B
  float*    invM = (float*)(ws + 131072);                          //   8192 B
  _Float16* Ap   = (_Float16*)(ws + 139264);                       // 16 MB
  _Float16* B1p  = (_Float16*)(ws + 139264 + 16777216);            // ~1 MB
  _Float16* B2p  = (_Float16*)(ws + 139264 + 16777216 + 1048576);  // 1 MB

  k_znorm <<<T_TOK / 256, 256, 0, stream>>>(z, invZ);
  k_mnorm <<<250, 256, 0, stream>>>(mem, invM);
  k_packB1<<<125, 256, 0, stream>>>(mem, invM, B1p);
  k_packB2<<<128, 256, 0, stream>>>(mem, B2p);
  k_packA <<<MT, 256, 0, stream>>>(z, invZ, Ap);
  k_main  <<<MT, 256, 0, stream>>>(Ap, B1p, B2p, zout, what);
}
